// TraceableFlowLMStep_66709432041828
// MI455X (gfx1250) — matrix operand order repaired and checked by reading, not compiled, not run on hardware
//
#include <hip/hip_runtime.h>
#include <math.h>

#define NLAY 6
#define BB 8
#define TT 4
#define LL 2048
#define HH 16
#define DD 64
#define EE 1024
#define FFN 4096
#define INW 32

typedef __attribute__((ext_vector_type(16))) _Float16 v16h;
typedef __attribute__((ext_vector_type(8)))  float    v8f;
typedef __attribute__((ext_vector_type(4)))  unsigned int v4u;
typedef __attribute__((ext_vector_type(8)))  int      v8i_;
typedef __attribute__((ext_vector_type(4)))  int      v4i_;

#define NEG_BIG (-3.4e38f)

// TDM staging geometry for attention: 128 k-rows of 64 f32, LDS row stride 68
#define SCHUNK 128
#define KSTRIDE 68

// ---------------------------------------------------------------------------
// Kernel 1: stream the whole KV cache input -> output, replacing NaN with 0.
// HBM-bound bulk (~1.6 GB); per-layer slice (128 MB) stays L2-resident for
// the attention kernel that follows it on the stream.
// ---------------------------------------------------------------------------
__global__ void cache_clean_copy(const float4* __restrict__ in,
                                 float4* __restrict__ out, long n4) {
  long i = (long)blockIdx.x * blockDim.x + threadIdx.x;
  long stride = (long)gridDim.x * blockDim.x;
  for (; i < n4; i += stride) {
    __builtin_prefetch(in + i + stride, 0, 0);   // global_prefetch_b8
    float4 v = in[i];
    v.x = isnan(v.x) ? 0.f : v.x;
    v.y = isnan(v.y) ? 0.f : v.y;
    v.z = isnan(v.z) ? 0.f : v.z;
    v.w = isnan(v.w) ? 0.f : v.w;
    out[i] = v;
  }
}

// ---------------------------------------------------------------------------
// Kernel 2: input projection  x[r,e] = sum_i clean(seq[r,i]) * in_w[e,i]
// ---------------------------------------------------------------------------
__global__ void input_proj(const float* __restrict__ seq,
                           const float* __restrict__ bos,
                           const float* __restrict__ in_w,
                           float* __restrict__ X) {
  int idx = blockIdx.x * 256 + threadIdx.x;        // 32*1024 outputs
  int r = idx / EE, e = idx % EE;
  const float* srow = seq + r * INW;
  const float* wrow = in_w + e * INW;
  float acc = 0.f;
#pragma unroll
  for (int i = 0; i < INW; ++i) {
    float s = srow[i];
    if (isnan(s)) s = bos[i];
    acc += s * wrow[i];
  }
  X[idx] = acc;
}

// ---------------------------------------------------------------------------
// Kernel 3: layernorm over E=1024 for 32 rows (one block per row)
// ---------------------------------------------------------------------------
__global__ void layernorm32(const float* __restrict__ X,
                            const float* __restrict__ w,
                            const float* __restrict__ b,
                            float* __restrict__ Y) {
  int row = blockIdx.x;
  int tid = threadIdx.x;
  const float* x = X + (long)row * EE;
  __shared__ float red[256];
  __shared__ float mean_s, rstd_s;
  float s = 0.f;
  for (int e = tid; e < EE; e += 256) s += x[e];
  red[tid] = s; __syncthreads();
  for (int o = 128; o > 0; o >>= 1) {
    if (tid < o) red[tid] += red[tid + o];
    __syncthreads();
  }
  if (tid == 0) mean_s = red[0] * (1.0f / EE);
  __syncthreads();
  float mean = mean_s;
  float v = 0.f;
  for (int e = tid; e < EE; e += 256) { float d = x[e] - mean; v += d * d; }
  red[tid] = v; __syncthreads();
  for (int o = 128; o > 0; o >>= 1) {
    if (tid < o) red[tid] += red[tid + o];
    __syncthreads();
  }
  if (tid == 0) rstd_s = rsqrtf(red[0] * (1.0f / EE) + 1e-5f);
  __syncthreads();
  float rstd = rstd_s;
  for (int e = tid; e < EE; e += 256)
    Y[(long)row * EE + e] = (x[e] - mean) * rstd * w[e] + b[e];
}

// ---------------------------------------------------------------------------
// Kernel 4: skinny WMMA GEMM  Y[32,N] = X[32,K] @ W[N,K]^T  (+res, +gelu)
// f32 -> f16 operands, f32 accumulation via v_wmma_f32_16x16x32_f16.
// Fragment layouts per CDNA5 ISA 7.12.2 (wave32).
// ---------------------------------------------------------------------------
__device__ inline float gelu_exact(float x) {
  return 0.5f * x * (1.0f + erff(x * 0.70710678118654752f));
}

template <int ACT>  // 0 = none, 1 = exact GELU
__global__ void gemm32_wmma(const float* __restrict__ X,
                            const float* __restrict__ W,
                            const float* __restrict__ Res,
                            float* __restrict__ Y, int Kd, int N) {
  int lane = threadIdx.x & 31;
  int wave = threadIdx.x >> 5;
  int n0 = blockIdx.x * 64 + wave * 16;
  int ln15 = lane & 15;
  bool hi = lane >= 16;

  const int aBase0 = hi ? 8 : 0;    // K origin of halves 0..7
  const int aBase1 = hi ? 24 : 16;  // K origin of halves 8..15
  const float* Xrow0 = X + (long)ln15 * Kd;         // rows 0..15 tile
  const float* Xrow1 = X + (long)(ln15 + 16) * Kd;  // rows 16..31 tile
  const float* Wrow = W + (long)(n0 + ln15) * Kd + (hi ? 8 : 0);

  v8f acc0 = {};
  v8f acc1 = {};

  for (int k0 = 0; k0 < Kd; k0 += 32) {
    v16h a0, a1, bf;
#pragma unroll
    for (int p = 0; p < 4; ++p) {
      float2 u = *(const float2*)(Xrow0 + k0 + aBase0 + 2 * p);
      a0[2 * p] = (_Float16)u.x; a0[2 * p + 1] = (_Float16)u.y;
      float2 v = *(const float2*)(Xrow0 + k0 + aBase1 + 2 * p);
      a0[8 + 2 * p] = (_Float16)v.x; a0[8 + 2 * p + 1] = (_Float16)v.y;
      float2 u1 = *(const float2*)(Xrow1 + k0 + aBase0 + 2 * p);
      a1[2 * p] = (_Float16)u1.x; a1[2 * p + 1] = (_Float16)u1.y;
      float2 v1 = *(const float2*)(Xrow1 + k0 + aBase1 + 2 * p);
      a1[8 + 2 * p] = (_Float16)v1.x; a1[8 + 2 * p + 1] = (_Float16)v1.y;
    }
#pragma unroll
    for (int p = 0; p < 8; ++p) {
      float2 wv = *(const float2*)(Wrow + k0 + 2 * p + (p >= 4 ? 8 : 0));
      bf[2 * p] = (_Float16)wv.x; bf[2 * p + 1] = (_Float16)wv.y;
    }
    acc0 = __builtin_amdgcn_wmma_f32_16x16x32_f16(false, a0, false, bf,
                                                  (short)0, acc0, false, false);
    acc1 = __builtin_amdgcn_wmma_f32_16x16x32_f16(false, a1, false, bf,
                                                  (short)0, acc1, false, false);
  }

  int col = n0 + ln15;
#pragma unroll
  for (int r = 0; r < 8; ++r) {
    int row0 = r + (hi ? 8 : 0);
    float y0 = acc0[r];
    float y1 = acc1[r];
    if (ACT == 1) { y0 = gelu_exact(y0); y1 = gelu_exact(y1); }
    if (Res) {
      y0 += Res[(long)row0 * N + col];
      y1 += Res[(long)(row0 + 16) * N + col];
    }
    Y[(long)row0 * N + col] = y0;
    Y[(long)(row0 + 16) * N + col] = y1;
  }
}

// ---------------------------------------------------------------------------
// Kernel 5: RoPE on q,k + scatter roped k and v into the cleaned output
// cache ring buffer at slot (pos+t) % L.  One thread per (b,t,h,pair).
// ---------------------------------------------------------------------------
__global__ void rope_scatter(const float* __restrict__ QKV,
                             const int* __restrict__ posL,
                             float* __restrict__ Qws,
                             float* __restrict__ cacheL) {
  int idx = blockIdx.x * 256 + threadIdx.x;  // B*T*H*(D/2) = 16384
  int j = idx & 31;
  int h = (idx >> 5) & 15;
  int t = (idx >> 9) & 3;
  int b = idx >> 11;
  int r = b * TT + t;
  int pos = posL[b];
  float freq = __expf((float)j * (-2.0f * 9.210340371976184f / (float)DD)); // ln(1e4)
  float ang = (float)(pos + t) * freq;
  float c = cosf(ang), s = sinf(ang);

  const float* q = QKV + (long)r * 3 * EE + h * DD + 2 * j;
  const float* k = q + EE;
  const float* v = q + 2 * EE;
  float qr = q[0], qi = q[1], kr = k[0], ki = k[1];

  Qws[(long)r * EE + h * DD + 2 * j]     = qr * c - qi * s;
  Qws[(long)r * EE + h * DD + 2 * j + 1] = qr * s + qi * c;

  int slot = (pos + t) % LL;
  float* kc = cacheL + ((long)b * LL + slot) * (HH * DD) + h * DD + 2 * j;
  float* vc = kc + (long)BB * LL * HH * DD;
  kc[0] = kr * c - ki * s;
  kc[1] = kr * s + ki * c;
  vc[0] = v[0];
  vc[1] = v[1];
}

// ---------------------------------------------------------------------------
// Kernel 6: attention for one (b,h) per block.  K-rows are staged into LDS
// by the Tensor Data Mover (tensor_load_to_lds, TENSORcnt) in 128-row tiles:
// 2D descriptor, tile 64 x 128 f32, tensor row stride 1024 elements, with
// TDM padding (4 DWORDs every 64 DWORDs) so LDS rows land at stride 68.
// One wave issues the DMA; all waves consume after s_wait_tensorcnt+barrier.
// ---------------------------------------------------------------------------
__global__ void attention(const float* __restrict__ Qws,
                          const float* __restrict__ cacheL,
                          const int* __restrict__ posL,
                          float* __restrict__ OUT) {
  int b = blockIdx.x / HH;
  int h = blockIdx.x % HH;
  int tid = threadIdx.x;

  __shared__ float kbuf[SCHUNK * KSTRIDE];  // 34.8 KB, rows 16B-aligned
  __shared__ float qs[TT][DD];
  __shared__ float sc[TT][1056];
  __shared__ float red[TT][256];
  __shared__ float Mx[TT], Sx[TT];

  int pos = posL[b];
  int smax = pos + TT;

  if (tid < TT * DD) {
    int t = tid / DD, d = tid % DD;
    qs[t][d] = Qws[(long)(b * TT + t) * EE + h * DD + d] * 0.125f; // 1/sqrt(64)
  }
  __syncthreads();

  // Static part of descriptor group 1 (see CDNA5 ISA):
  //   data_size=4B (2), pad_enable, pad_interval=64 DW (5), pad_amount=4 DW (3)
  //   tensor_dim0=64, tensor_dim1=2048, tile_dim0=64, tile_dim1=128
  //   tensor_dim0_stride=1024 elements
  v8i_ g1;
  g1[0] = (int)((2u << 16) | (1u << 20) | (5u << 22) | (3u << 25));
  g1[1] = (int)(64u << 16);            // tensor_dim0[15:0] -> bits 63:48
  g1[2] = (int)(2048u << 16);          // tensor_dim1[15:0] -> bits 95:80
  g1[3] = (int)(64u << 16);            // tile_dim0 -> bits 127:112
  g1[4] = (int)(SCHUNK);               // tile_dim1 -> bits 143:128
  g1[5] = (int)(1024);                 // tensor_dim0_stride[31:0]
  g1[6] = 0;
  g1[7] = 0;
  v4i_ gz4 = {0, 0, 0, 0};
  v8i_ gz8 = {0, 0, 0, 0, 0, 0, 0, 0};
  unsigned ldsOff = (unsigned)(unsigned long long)(uintptr_t)&kbuf[0];
  const float* Kc = cacheL + (long)b * LL * (HH * DD) + h * DD;

  int rrow = tid & (SCHUNK - 1);   // which staged row this thread scores
  int tp = tid >> 7;               // 0 -> t={0,1}, 1 -> t={2,3}
  float lmA = NEG_BIG, lmB = NEG_BIG;

  int nchunks = (smax + SCHUNK - 1) / SCHUNK;
  for (int c = 0; c < nchunks; ++c) {
    int s0 = c * SCHUNK;
    if (tid < 32) {
      unsigned long long ga = (unsigned long long)(uintptr_t)(Kc + (long)s0 * (HH * DD));
      v4u g0;
      g0[0] = 1u;                                   // count=1 valid descriptor
      g0[1] = ldsOff;                               // LDS destination
      g0[2] = (unsigned)(ga & 0xFFFFFFFFu);         // global_addr[31:0]
      g0[3] = (unsigned)((ga >> 32) & 0x1FFFFFFu) | (2u << 30);  // addr[56:32] | type=2
      __builtin_amdgcn_tensor_load_to_lds(g0, g1, gz4, gz4, gz8, 0);
      __builtin_amdgcn_s_wait_tensorcnt(0);
    }
    __syncthreads();

    int s = s0 + rrow;
    if (s < smax) {
      const float* kr = &kbuf[rrow * KSTRIDE];
      int t0 = 2 * tp, t1 = 2 * tp + 1;
      float d0 = 0.f, d1 = 0.f;
#pragma unroll
      for (int d = 0; d < DD; d += 4) {
        float4 kv = *(const float4*)(kr + d);
        d0 += qs[t0][d] * kv.x + qs[t0][d + 1] * kv.y + qs[t0][d + 2] * kv.z + qs[t0][d + 3] * kv.w;
        d1 += qs[t1][d] * kv.x + qs[t1][d + 1] * kv.y + qs[t1][d + 2] * kv.z + qs[t1][d + 3] * kv.w;
      }
      float sA = (s <= pos + t0) ? d0 : NEG_BIG; sc[t0][s] = sA; lmA = fmaxf(lmA, sA);
      float sB = (s <= pos + t1) ? d1 : NEG_BIG; sc[t1][s] = sB; lmB = fmaxf(lmB, sB);
    }
    __syncthreads();   // protect kbuf before next chunk's DMA overwrites it
  }

  // block-wide max per t
#pragma unroll
  for (int t = 0; t < TT; ++t)
    red[t][tid] = ((t >> 1) == tp) ? ((t & 1) ? lmB : lmA) : NEG_BIG;
  __syncthreads();
  for (int o = 128; o > 0; o >>= 1) {
    if (tid < o) {
#pragma unroll
      for (int t = 0; t < TT; ++t)
        red[t][tid] = fmaxf(red[t][tid], red[t][tid + o]);
    }
    __syncthreads();
  }
  if (tid < TT) Mx[tid] = red[tid][0];
  __syncthreads();

  // exponentiate + block-wide sum per t
  float ls0 = 0.f, ls1 = 0.f, ls2 = 0.f, ls3 = 0.f;
  for (int s = tid; s < smax; s += 256) {
    float p0 = __expf(sc[0][s] - Mx[0]); sc[0][s] = p0; ls0 += p0;
    float p1 = __expf(sc[1][s] - Mx[1]); sc[1][s] = p1; ls1 += p1;
    float p2 = __expf(sc[2][s] - Mx[2]); sc[2][s] = p2; ls2 += p2;
    float p3 = __expf(sc[3][s] - Mx[3]); sc[3][s] = p3; ls3 += p3;
  }
  red[0][tid] = ls0; red[1][tid] = ls1; red[2][tid] = ls2; red[3][tid] = ls3;
  __syncthreads();
  for (int o = 128; o > 0; o >>= 1) {
    if (tid < o) {
#pragma unroll
      for (int t = 0; t < TT; ++t)
        red[t][tid] += red[t][tid + o];
    }
    __syncthreads();
  }
  if (tid < TT) Sx[tid] = red[tid][0];
  __syncthreads();

  // pass 2: thread = (t, d); v reads are naturally coalesced (d contiguous)
  int t = tid >> 6;
  int d = tid & 63;
  const float* Vc = cacheL + ((long)BB * LL + (long)b * LL) * (HH * DD) + h * DD + d;
  float acc = 0.f;
  for (int s = 0; s < smax; ++s) acc += sc[t][s] * Vc[(long)s * (HH * DD)];
  OUT[(long)(b * TT + t) * EE + h * DD + d] = acc / Sx[t];
}

// ---------------------------------------------------------------------------
// Kernel 7: EOS head  is_eos[r] = x_final[r,:] . eos_w + eos_b
// ---------------------------------------------------------------------------
__global__ void eos_head(const float* __restrict__ Xf,
                         const float* __restrict__ ew,
                         const float* __restrict__ eb,
                         float* __restrict__ out) {
  int r = blockIdx.x;
  int tid = threadIdx.x;
  __shared__ float red[256];
  float s = 0.f;
  for (int e = tid; e < EE; e += 256) s += Xf[(long)r * EE + e] * ew[e];
  red[tid] = s; __syncthreads();
  for (int o = 128; o > 0; o >>= 1) {
    if (tid < o) red[tid] += red[tid + o];
    __syncthreads();
  }
  if (tid == 0) out[r] = red[0] + eb[0];
}

__global__ void write_positions(const int* __restrict__ pos, float* __restrict__ out) {
  int idx = blockIdx.x * blockDim.x + threadIdx.x;
  if (idx < NLAY * BB) out[idx] = (float)(pos[idx] + TT);
}

// ---------------------------------------------------------------------------
extern "C" void kernel_launch(void* const* d_in, const int* in_sizes, int n_in,
                              void* d_out, int out_size, void* d_ws, size_t ws_size,
                              hipStream_t stream) {
  const float* seq   = (const float*)d_in[0];
  const float* bos   = (const float*)d_in[1];
  const float* caches = (const float*)d_in[2];
  const int*   positions = (const int*)d_in[3];
  const float* in_w  = (const float*)d_in[4];
  const float* ip_w  = (const float*)d_in[5];
  const float* op_w  = (const float*)d_in[6];
  const float* n1_w  = (const float*)d_in[7];
  const float* n1_b  = (const float*)d_in[8];
  const float* n2_w  = (const float*)d_in[9];
  const float* n2_b  = (const float*)d_in[10];
  const float* l1_w  = (const float*)d_in[11];
  const float* l2_w  = (const float*)d_in[12];
  const float* on_w  = (const float*)d_in[13];
  const float* on_b  = (const float*)d_in[14];
  const float* eos_w = (const float*)d_in[15];
  const float* eos_b = (const float*)d_in[16];

  const long cacheElems = (long)NLAY * 2 * BB * LL * HH * DD;  // 201326592
  float* out = (float*)d_out;
  float* outX = out;                    // 32768
  float* outEos = out + 32768;          // 32
  float* outCache = out + 32800;        // cacheElems
  float* outPos = out + 32800 + cacheElems;  // 48

  float* ws  = (float*)d_ws;
  float* X   = ws;                 // 32*1024
  float* XN  = ws + 32768;         // 32*1024
  float* QKV = ws + 65536;         // 32*3072
  float* Qb  = ws + 163840;        // 32*1024 (roped q)
  float* Hb  = ws + 196608;        // 32*4096
  float* ATT = ws + 327680;        // 32*1024

  cache_clean_copy<<<8192, 256, 0, stream>>>((const float4*)caches,
                                             (float4*)outCache, cacheElems / 4);
  input_proj<<<128, 256, 0, stream>>>(seq, bos, in_w, X);

  for (int i = 0; i < NLAY; ++i) {
    float* cacheL = outCache + (long)i * 2 * BB * LL * HH * DD;
    const int* posL = positions + i * BB;

    layernorm32<<<32, 256, 0, stream>>>(X, n1_w + i * EE, n1_b + i * EE, XN);
    gemm32_wmma<0><<<(3 * EE) / 64, 128, 0, stream>>>(
        XN, ip_w + (long)i * 3 * EE * EE, nullptr, QKV, EE, 3 * EE);
    rope_scatter<<<64, 256, 0, stream>>>(QKV, posL, Qb, cacheL);
    attention<<<BB * HH, 256, 0, stream>>>(Qb, cacheL, posL, ATT);
    gemm32_wmma<0><<<EE / 64, 128, 0, stream>>>(
        ATT, op_w + (long)i * EE * EE, X, X, EE, EE);
    layernorm32<<<32, 256, 0, stream>>>(X, n2_w + i * EE, n2_b + i * EE, XN);
    gemm32_wmma<1><<<FFN / 64, 128, 0, stream>>>(
        XN, l1_w + (long)i * FFN * EE, nullptr, Hb, EE, FFN);
    gemm32_wmma<0><<<EE / 64, 128, 0, stream>>>(
        Hb, l2_w + (long)i * EE * FFN, X, X, FFN, EE);
  }

  layernorm32<<<32, 256, 0, stream>>>(X, on_w, on_b, outX);
  eos_head<<<32, 256, 0, stream>>>(outX, eos_w, eos_b, outEos);
  write_positions<<<1, 64, 0, stream>>>(positions, outPos);
}
